// MambaBranch_44272522887547
// MI455X (gfx1250) — matrix operand order repaired and checked by reading, not compiled, not run on hardware
//
#include <hip/hip_runtime.h>
#include <hip/hip_bf16.h>

// ---------------------------------------------------------------------------
// Mamba-branch classifier for MI455X (gfx1250).
// All large GEMMs run on V_WMMA_F32_16X16X32_F16 (f16 in, f32 acc).
// Wave tile 64x32 (4x2 WMMA accumulators) for higher WMMA-per-load density.
// ---------------------------------------------------------------------------

typedef __attribute__((ext_vector_type(16))) _Float16 v16h;
typedef __attribute__((ext_vector_type(8)))  _Float16 v8h;
typedef __attribute__((ext_vector_type(8)))  float    v8f;
typedef __attribute__((ext_vector_type(4)))  float    v4f;

#define NB      16384
#define D_IN    512
#define D_MODEL 256
#define D_INNER 512
#define D_STATE 16
#define DT_RANK 16
#define NBLK    5
#define NCLS    1000

// ---------------------------------------------------------------------------
// f32 -> f16 conversion, 8 elements per thread (vectorized)
// ---------------------------------------------------------------------------
__global__ __launch_bounds__(256) void cvt_f32_f16(const float* __restrict__ src,
                                                   _Float16* __restrict__ dst, int n) {
  int i = (blockIdx.x * 256 + threadIdx.x) * 8;
  if (i + 7 < n) {
    v4f a = *(const v4f*)(src + i);
    v4f b = *(const v4f*)(src + i + 4);
    v8h o;
#pragma unroll
    for (int j = 0; j < 4; ++j) { o[j] = (_Float16)a[j]; o[j + 4] = (_Float16)b[j]; }
    *(v8h*)(dst + i) = o;
  } else {
    for (int j = i; j < n; ++j) dst[j] = (_Float16)src[j];
  }
}

__device__ __forceinline__ float siluf(float x) {
  return x / (1.0f + __expf(-x));
}
__device__ __forceinline__ float softplusf(float x) {
  return (x > 20.0f) ? x : log1pf(__expf(x));
}

// ---------------------------------------------------------------------------
// WMMA fragment loads (wave32 layouts per CDNA5 ISA 7.12.2)
// A: 16x32 f16 tile of row-major A[M,K]. Lane L: row = L%16,
//    elems 0..7 = K kb..kb+7, elems 8..15 = K kb+16..kb+23, kb = 8*(L>=16).
// B: 32x16 f16 tile (K x N) sourced from row-major W[N,K] (B = W^T).
//    Lane L: col = L%16, same K order as A: elems 0..7 = K kb..kb+7, 8..15 = K kb+16..kb+23.
// ---------------------------------------------------------------------------
__device__ __forceinline__ v16h load_a_frag(const _Float16* __restrict__ A, int lda,
                                            int row, int k0, int lane) {
  int kb = k0 + ((lane & 16) ? 8 : 0);
  const _Float16* p = A + (size_t)row * lda + kb;
  v8h lo = *(const v8h*)p;          // K = kb .. kb+7
  v8h hi = *(const v8h*)(p + 16);   // K = kb+16 .. kb+23
  v16h a;
#pragma unroll
  for (int i = 0; i < 8; ++i) { a[i] = lo[i]; a[i + 8] = hi[i]; }
  return a;
}

__device__ __forceinline__ v16h load_b_frag(const _Float16* __restrict__ W, int ldw,
                                            int col, int k0, int lane) {
  int kb = k0 + ((lane & 16) ? 8 : 0);
  const _Float16* p = W + (size_t)col * ldw + kb;
  v8h lo = *(const v8h*)p;
  v8h hi = *(const v8h*)(p + 16);
  v16h b;
#pragma unroll
  for (int i = 0; i < 8; ++i) { b[i] = lo[i]; b[i + 8] = hi[i]; }
  return b;
}

// ---------------------------------------------------------------------------
// Generic WMMA GEMM: C[M,N] = A[M,K] @ W[N,K]^T, f32 accumulate.
// Block tile 128(M) x 128(N): 8 waves (2 along M x 4 along N),
// each wave computes 64x32 via a 4x2 grid of WMMA accumulators.
// MODE 0: Cf = acc (+ bias if non-null)          [dbl / logits]
// MODE 1: in_proj epilogue -> xc (SiLU conv tap) and sz (SiLU z), f16
// MODE 2: residual: Cf += acc; Ch = (f16)Cf      [out_proj]
// MODE 3: Cf = acc + bias; Ch = (f16)Cf          [pre projection]
// ---------------------------------------------------------------------------
template <int MODE>
__global__ __launch_bounds__(256) void gemm_wmma(
    const _Float16* __restrict__ A, int lda,
    const _Float16* __restrict__ W, int ldw,
    int M, int N, int K,
    const float* __restrict__ bias,
    float* __restrict__ Cf,
    _Float16* __restrict__ Ch,
    const float* __restrict__ convw,   // MODE 1: conv_w block base [512,4]
    const float* __restrict__ convb,   // MODE 1: conv_b block base [512]
    _Float16* __restrict__ xc_out,     // MODE 1
    _Float16* __restrict__ sz_out) {   // MODE 1
  const int lane = threadIdx.x & 31;
  const int wave = threadIdx.x >> 5;
  const int wm = wave & 1;            // 2 waves along M, 64 rows each
  const int wn = wave >> 1;           // 4 waves along N, 32 cols each
  const int m0 = blockIdx.x * 128 + wm * 64;
  const int n0 = blockIdx.y * 128 + wn * 32;

  v8f acc[4][2] = {};

  int rowA[4];
#pragma unroll
  for (int t = 0; t < 4; ++t) rowA[t] = m0 + 16 * t + (lane & 15);
  int c0 = n0 + (lane & 15);       if (c0 >= N) c0 = N - 1;   // clamp OOB loads
  int c1 = n0 + 16 + (lane & 15);  if (c1 >= N) c1 = N - 1;

  for (int k0 = 0; k0 < K; k0 += 32) {
    v16h b0 = load_b_frag(W, ldw, c0, k0, lane);
    v16h b1 = load_b_frag(W, ldw, c1, k0, lane);
    v16h a0 = load_a_frag(A, lda, rowA[0], k0, lane);
    v16h a1 = load_a_frag(A, lda, rowA[1], k0, lane);
    v16h a2 = load_a_frag(A, lda, rowA[2], k0, lane);
    v16h a3 = load_a_frag(A, lda, rowA[3], k0, lane);
    acc[0][0] = __builtin_amdgcn_wmma_f32_16x16x32_f16(false, a0, false, b0, (short)0, acc[0][0], false, false);
    acc[0][1] = __builtin_amdgcn_wmma_f32_16x16x32_f16(false, a0, false, b1, (short)0, acc[0][1], false, false);
    acc[1][0] = __builtin_amdgcn_wmma_f32_16x16x32_f16(false, a1, false, b0, (short)0, acc[1][0], false, false);
    acc[1][1] = __builtin_amdgcn_wmma_f32_16x16x32_f16(false, a1, false, b1, (short)0, acc[1][1], false, false);
    acc[2][0] = __builtin_amdgcn_wmma_f32_16x16x32_f16(false, a2, false, b0, (short)0, acc[2][0], false, false);
    acc[2][1] = __builtin_amdgcn_wmma_f32_16x16x32_f16(false, a2, false, b1, (short)0, acc[2][1], false, false);
    acc[3][0] = __builtin_amdgcn_wmma_f32_16x16x32_f16(false, a3, false, b0, (short)0, acc[3][0], false, false);
    acc[3][1] = __builtin_amdgcn_wmma_f32_16x16x32_f16(false, a3, false, b1, (short)0, acc[3][1], false, false);
  }

  // Epilogue. C/D layout: lane L -> col = n0t + L%16; VGPR r -> row = m0t + r + 8*(L>=16).
  const int rofs = (lane & 16) ? 8 : 0;
#pragma unroll
  for (int tm = 0; tm < 4; ++tm) {
#pragma unroll
    for (int tn = 0; tn < 2; ++tn) {
      int col = n0 + tn * 16 + (lane & 15);
      if (col >= N) continue;
#pragma unroll
      for (int r = 0; r < 8; ++r) {
        int row = m0 + tm * 16 + r + rofs;
        float v = acc[tm][tn][r];
        if (MODE == 0) {
          float o = v + (bias ? bias[col] : 0.0f);
          Cf[(size_t)row * N + col] = o;
        } else if (MODE == 1) {
          if (col < D_INNER) {
            float t = v * convw[col * 4 + 3] + convb[col];
            xc_out[(size_t)row * D_INNER + col] = (_Float16)siluf(t);
          } else {
            sz_out[(size_t)row * D_INNER + (col - D_INNER)] = (_Float16)siluf(v);
          }
        } else if (MODE == 2) {
          size_t idx = (size_t)row * N + col;
          float o = Cf[idx] + v;
          Cf[idx] = o;
          Ch[idx] = (_Float16)o;
        } else {  // MODE 3
          size_t idx = (size_t)row * N + col;
          float o = v + bias[col];
          Cf[idx] = o;
          Ch[idx] = (_Float16)o;
        }
      }
    }
  }
}

// ---------------------------------------------------------------------------
// Per-row SSM gating: one block per batch row.
// dbl[b,0:16]=dt_in, [16:32]=Bc, [32:48]=Cc.
// bc = <Bc,Cc>; dt[k] = softplus(<dt_in, dtw[k,:]> + dtb[k]);
// y[k] = (dt*xc*bc + D*xc) * sz    -> f16 for out_proj GEMM.
// ---------------------------------------------------------------------------
__global__ __launch_bounds__(256) void y_kernel(
    const float* __restrict__ dbl,      // [NB,48]
    const float* __restrict__ dtw,      // [512,16]
    const float* __restrict__ dtb,      // [512]
    const float* __restrict__ Dk,       // [512]
    const _Float16* __restrict__ xc,    // [NB,512]
    const _Float16* __restrict__ sz,    // [NB,512]
    _Float16* __restrict__ y) {         // [NB,512]
  const int b = blockIdx.x;
  const int t = threadIdx.x;
  __shared__ float s[48];
  if (t < 48) s[t] = dbl[(size_t)b * 48 + t];
  __syncthreads();
  float bc = 0.0f;
#pragma unroll
  for (int j = 0; j < D_STATE; ++j) bc += s[DT_RANK + j] * s[DT_RANK + D_STATE + j];
#pragma unroll
  for (int kk = 0; kk < 2; ++kk) {
    int k = t + kk * 256;
    float d = dtb[k];
#pragma unroll
    for (int j = 0; j < DT_RANK; ++j) d += s[j] * dtw[k * DT_RANK + j];
    float dt = softplusf(d);
    float xcv = (float)xc[(size_t)b * D_INNER + k];
    float yv = (dt * xcv * bc + Dk[k] * xcv) * (float)sz[(size_t)b * D_INNER + k];
    y[(size_t)b * D_INNER + k] = (_Float16)yv;
  }
}

// ---------------------------------------------------------------------------
// LayerNorm over rows of 256, one block per row (wave32 reductions).
// ---------------------------------------------------------------------------
__global__ __launch_bounds__(256) void ln_kernel(
    const float* __restrict__ h, const float* __restrict__ g,
    const float* __restrict__ bta, _Float16* __restrict__ hn) {
  const int row = blockIdx.x;
  const int t = threadIdx.x;
  float v = h[(size_t)row * D_MODEL + t];
  float s1 = v, s2 = v * v;
#pragma unroll
  for (int o = 16; o > 0; o >>= 1) {
    s1 += __shfl_xor(s1, o, 32);
    s2 += __shfl_xor(s2, o, 32);
  }
  __shared__ float sm1[8], sm2[8];
  if ((t & 31) == 0) { sm1[t >> 5] = s1; sm2[t >> 5] = s2; }
  __syncthreads();
  float t1 = 0.0f, t2 = 0.0f;
#pragma unroll
  for (int i = 0; i < 8; ++i) { t1 += sm1[i]; t2 += sm2[i]; }
  float mu = t1 * (1.0f / D_MODEL);
  float var = t2 * (1.0f / D_MODEL) - mu * mu;
  float inv = rsqrtf(var + 1e-5f);
  hn[(size_t)row * D_MODEL + t] = (_Float16)((v - mu) * inv * g[t] + bta[t]);
}

// ---------------------------------------------------------------------------
// Host orchestration
// ---------------------------------------------------------------------------
static inline dim3 gemm_grid(int M, int N) { return dim3(M / 128, (N + 127) / 128); }

extern "C" void kernel_launch(void* const* d_in, const int* in_sizes, int n_in,
                              void* d_out, int out_size, void* d_ws, size_t ws_size,
                              hipStream_t stream) {
  (void)in_sizes; (void)n_in; (void)out_size; (void)ws_size;
  const float* x      = (const float*)d_in[0];
  const float* pre_w  = (const float*)d_in[1];
  const float* pre_b  = (const float*)d_in[2];
  const float* in_w   = (const float*)d_in[3];
  const float* conv_w = (const float*)d_in[4];
  const float* conv_b = (const float*)d_in[5];
  const float* xp_w   = (const float*)d_in[6];
  const float* dt_w   = (const float*)d_in[7];
  const float* dt_b   = (const float*)d_in[8];
  /* d_in[9] = A_log: unused by the reference math */
  const float* Dk     = (const float*)d_in[10];
  const float* out_w  = (const float*)d_in[11];
  const float* ln_g   = (const float*)d_in[12];
  const float* ln_b   = (const float*)d_in[13];
  const float* cls_w  = (const float*)d_in[14];
  const float* cls_b  = (const float*)d_in[15];
  float* out = (float*)d_out;

  char* ws = (char*)d_ws;
  size_t off = 0;
  auto alloc = [&](size_t bytes) -> void* {
    void* p = (void*)(ws + off);
    off += (bytes + 255) & ~(size_t)255;
    return p;
  };

  _Float16* x16   = (_Float16*)alloc((size_t)NB * D_IN * 2);
  float*    h32   = (float*)   alloc((size_t)NB * D_MODEL * 4);
  _Float16* h16   = (_Float16*)alloc((size_t)NB * D_MODEL * 2);
  _Float16* xc16  = (_Float16*)alloc((size_t)NB * D_INNER * 2);
  _Float16* sz16  = (_Float16*)alloc((size_t)NB * D_INNER * 2);
  _Float16* y16   = (_Float16*)alloc((size_t)NB * D_INNER * 2);
  float*    dbl   = (float*)   alloc((size_t)NB * 48 * 4);
  _Float16* hn16  = (_Float16*)alloc((size_t)NB * D_MODEL * 2);
  _Float16* pre16 = (_Float16*)alloc((size_t)D_MODEL * D_IN * 2);
  _Float16* in16  = (_Float16*)alloc((size_t)NBLK * 2 * D_INNER * D_MODEL * 2);
  _Float16* xp16  = (_Float16*)alloc((size_t)NBLK * 48 * D_INNER * 2);
  _Float16* op16  = (_Float16*)alloc((size_t)NBLK * D_MODEL * D_INNER * 2);
  _Float16* cls16 = (_Float16*)alloc((size_t)NCLS * D_MODEL * 2);

  auto cvt = [&](const float* s, _Float16* d, int n) {
    int threads = (n + 7) / 8;
    cvt_f32_f16<<<(threads + 255) / 256, 256, 0, stream>>>(s, d, n);
  };
  cvt(x,     x16,   NB * D_IN);
  cvt(pre_w, pre16, D_MODEL * D_IN);
  cvt(in_w,  in16,  NBLK * 2 * D_INNER * D_MODEL);
  cvt(xp_w,  xp16,  NBLK * 48 * D_INNER);
  cvt(out_w, op16,  NBLK * D_MODEL * D_INNER);
  cvt(cls_w, cls16, NCLS * D_MODEL);

  // h = x @ pre_w.T + pre_b   (M=16384, N=256, K=512)
  gemm_wmma<3><<<gemm_grid(NB, D_MODEL), 256, 0, stream>>>(
      x16, D_IN, pre16, D_IN, NB, D_MODEL, D_IN,
      pre_b, h32, h16, nullptr, nullptr, nullptr, nullptr);

  for (int i = 0; i < NBLK; ++i) {
    // xz = h @ in_proj.T, fused SiLU epilogue -> xc16, sz16 (N=1024, K=256)
    gemm_wmma<1><<<gemm_grid(NB, 2 * D_INNER), 256, 0, stream>>>(
        h16, D_MODEL, in16 + (size_t)i * 2 * D_INNER * D_MODEL, D_MODEL,
        NB, 2 * D_INNER, D_MODEL,
        nullptr, nullptr, nullptr,
        conv_w + (size_t)i * D_INNER * 4, conv_b + (size_t)i * D_INNER,
        xc16, sz16);

    // dbl = xc @ x_proj.T  (N=48, K=512)
    gemm_wmma<0><<<gemm_grid(NB, 48), 256, 0, stream>>>(
        xc16, D_INNER, xp16 + (size_t)i * 48 * D_INNER, D_INNER,
        NB, 48, D_INNER,
        nullptr, dbl, nullptr, nullptr, nullptr, nullptr, nullptr);

    // per-row gating -> y16
    y_kernel<<<NB, 256, 0, stream>>>(
        dbl, dt_w + (size_t)i * D_INNER * DT_RANK, dt_b + (size_t)i * D_INNER,
        Dk + (size_t)i * D_INNER, xc16, sz16, y16);

    // h += y @ out_proj.T  (N=256, K=512), refresh h16
    gemm_wmma<2><<<gemm_grid(NB, D_MODEL), 256, 0, stream>>>(
        y16, D_INNER, op16 + (size_t)i * D_MODEL * D_INNER, D_INNER,
        NB, D_MODEL, D_INNER,
        nullptr, h32, h16, nullptr, nullptr, nullptr, nullptr);
  }

  // LayerNorm -> hn16
  ln_kernel<<<NB, 256, 0, stream>>>(h32, ln_g, ln_b, hn16);

  // logits = hn @ cls_w.T + cls_b  (N=1000, K=256)
  gemm_wmma<0><<<gemm_grid(NB, NCLS), 256, 0, stream>>>(
      hn16, D_MODEL, cls16, D_MODEL, NB, NCLS, D_MODEL,
      cls_b, out, nullptr, nullptr, nullptr, nullptr, nullptr);
}
